// BiMambaBlock_4544075399352
// MI455X (gfx1250) — matrix operand order repaired and checked by reading, not compiled, not run on hardware
//
#include <hip/hip_runtime.h>

// ---------------------------------------------------------------------------
// BiMamba block for MI455X (gfx1250, wave32, WMMA).
// GEMMs: bf16 v_wmma_f32_16x16x32_bf16, fp32 accumulate, 2x2 tiles per wave,
// LDS double-buffered with register prefetch (1 barrier per K-step).
// All GEMM indexing is 32-bit to enable SADDR+voffset (GVS) addressing.
// ---------------------------------------------------------------------------

#define B_    8
#define L_    1024
#define DM_   256
#define DI_   512
#define DS_   16
#define DTR_  16
#define XDBL_ 48          // DTR + 2*DS
#define ROWS_ (B_ * L_)   // 8192

typedef __attribute__((ext_vector_type(16))) __bf16 v16bf;
typedef __attribute__((ext_vector_type(8)))  __bf16 v8bf;
typedef __attribute__((ext_vector_type(8)))  float  v8f;

__device__ __forceinline__ float silu_f(float v) {
    return v / (1.f + __expf(-v));
}
__device__ __forceinline__ float softplus_f(float v) {
    return (v > 20.f) ? v : log1pf(__expf(v));
}

// ---------------------------------------------------------------------------
// LayerNorm over last dim (256). One wave per row, 8 waves per block.
// ---------------------------------------------------------------------------
__global__ void ln_kernel(const float* __restrict__ x,
                          const float* __restrict__ g,
                          const float* __restrict__ b,
                          float* __restrict__ xn) {
    const int wave = threadIdx.x >> 5;
    const int lane = threadIdx.x & 31;
    const int row  = blockIdx.x * 8 + wave;
    if (row >= ROWS_) return;
    const float* xr = x + (size_t)row * DM_;

    float v[8];
    float s = 0.f;
#pragma unroll
    for (int i = 0; i < 8; ++i) { v[i] = xr[lane + 32 * i]; s += v[i]; }
#pragma unroll
    for (int off = 16; off >= 1; off >>= 1) s += __shfl_xor(s, off, 32);
    const float mu = s * (1.f / DM_);

    float var = 0.f;
#pragma unroll
    for (int i = 0; i < 8; ++i) { float d = v[i] - mu; var += d * d; }
#pragma unroll
    for (int off = 16; off >= 1; off >>= 1) var += __shfl_xor(var, off, 32);
    const float rs = rsqrtf(var * (1.f / DM_) + 1e-5f);

    float* o = xn + (size_t)row * DM_;
#pragma unroll
    for (int i = 0; i < 8; ++i) {
        const int c = lane + 32 * i;
        o[c] = (v[i] - mu) * rs * g[c] + b[c];
    }
}

// ---------------------------------------------------------------------------
// bf16 WMMA GEMM:  C[M,N] = act(A[M,K] * W[K,N] + bias)
//   KK     : compile-time K (16 is zero-padded to one 32-wide WMMA step)
//   NBOUND : true when N is not a multiple of the 64-col block tile
// Block: 256 threads = 8 waves (4 x 2) -> 128x64 tile; each wave owns a
// 32x32 subtile (2x2 accumulators -> 4 wmma per 32-K step).
// LDS (double-buffered) holds the WMMA lane layout:
//   Asw[buf][(row>>4)*2+khalf][row&15][e]  -> lane reads 32 contiguous bytes
//   Bt [buf][n_local][k]                   -> lane reads 32 contiguous bytes
// act: 0 = none, 1 = softplus
// ---------------------------------------------------------------------------
template <int KK, bool NBOUND>
__global__ __launch_bounds__(256)
void gemm_wmma_bf16(const float* __restrict__ A, int lda,
                    const float* __restrict__ W, int ldb,
                    float* __restrict__ C, int ldc,
                    int M, int N,
                    const float* __restrict__ bias, int act) {
    constexpr bool KB     = (KK % 32) != 0;      // ragged K (only KK==16 here)
    constexpr int  KPAD   = ((KK + 31) / 32) * 32;
    constexpr int  NSTEPS = KPAD / 32;

    __shared__ __bf16 Asw[2][16][16][16];   // 2 x 8 KB
    __shared__ __bf16 Bt[2][64][32];        // 2 x 4 KB

    const int tid  = threadIdx.x;
    const int m0   = blockIdx.x * 128;
    const int n0   = blockIdx.y * 64;
    const int wave = tid >> 5;
    const int lane = tid & 31;
    const int wm   = wave & 3;      // 0..3 : 32-row band
    const int wn   = wave >> 2;     // 0..1 : 32-col band
    const int half = lane >> 4;
    const int lr   = lane & 15;

    // Staging coordinates (fixed per thread).
    const int ar  = tid >> 2;               // A rows: ar and ar+64
    const int ac0 = (tid & 3) * 8;          // A k-offset: 0,8,16,24
    const int akh = (ac0 >> 3) & 1;         // k half (0: k%16<8)
    const int aeg = (ac0 >> 4) & 1;         // element group (k<16 vs k>=16)
    const int bn  = tid & 63;               // B n_local: 0..63
    const int bc0 = (tid >> 6) * 8;         // B k-offset: 0,8,16,24

    // 32-bit base offsets (all matrices here are < 2^31 elements).
    const int aoff0 = (m0 + ar) * lda + ac0;        // row ar
    const int aoff1 = (m0 + ar + 64) * lda + ac0;   // row ar+64
    const int nG    = n0 + bn;
    const int nCl   = NBOUND ? ((nG < N) ? nG : (N - 1)) : nG;

    // Prefetch registers.
    float a0[8], a1[8], bb[8];

    auto loadA = [&](int k0) {
        if (KB) {
#pragma unroll
            for (int j = 0; j < 8; ++j) {
                const int k  = k0 + ac0 + j;
                const int kc = (k < KK) ? k : (KK - 1);
                const int o  = kc - ac0;     // relative to aoff base
                const float v0 = A[aoff0 + o];
                const float v1 = A[aoff1 + o];
                a0[j] = (k < KK) ? v0 : 0.f;
                a1[j] = (k < KK) ? v1 : 0.f;
            }
        } else {
            const float* g0 = A + aoff0 + k0;
            const float* g1 = A + aoff1 + k0;
            const float4 f0 = *(const float4*)(g0);
            const float4 f1 = *(const float4*)(g0 + 4);
            const float4 f2 = *(const float4*)(g1);
            const float4 f3 = *(const float4*)(g1 + 4);
            a0[0] = f0.x; a0[1] = f0.y; a0[2] = f0.z; a0[3] = f0.w;
            a0[4] = f1.x; a0[5] = f1.y; a0[6] = f1.z; a0[7] = f1.w;
            a1[0] = f2.x; a1[1] = f2.y; a1[2] = f2.z; a1[3] = f2.w;
            a1[4] = f3.x; a1[5] = f3.y; a1[6] = f3.z; a1[7] = f3.w;
        }
    };
    auto loadB = [&](int k0) {
#pragma unroll
        for (int j = 0; j < 8; ++j) {
            const int k  = k0 + bc0 + j;
            const int kc = KB ? ((k < KK) ? k : (KK - 1)) : k;
            float v = W[kc * ldb + nCl];     // 32-bit offset -> GVS addressing
            if (KB && k >= KK) v = 0.f;
            if (NBOUND && nG >= N) v = 0.f;
            bb[j] = v;
        }
    };
    auto stage = [&](int buf) {
        v8bf o0, o1, ob;
#pragma unroll
        for (int j = 0; j < 8; ++j) {
            o0[j] = (__bf16)a0[j];
            o1[j] = (__bf16)a1[j];
            ob[j] = (__bf16)bb[j];
        }
        *(v8bf*)&Asw[buf][((ar >> 4) * 2) + akh][ar & 15][aeg * 8]        = o0;
        *(v8bf*)&Asw[buf][(((ar + 64) >> 4) * 2) + akh][ar & 15][aeg * 8] = o1;
        *(v8bf*)&Bt[buf][bn][((bc0 >> 3) & 1) * 16 + ((bc0 >> 4) & 1) * 8] = ob;
    };

    v8f acc[2][2];
#pragma unroll
    for (int mi = 0; mi < 2; ++mi)
#pragma unroll
        for (int ni = 0; ni < 2; ++ni) acc[mi][ni] = (v8f){};

    loadA(0);
    loadB(0);
    stage(0);
    __syncthreads();

    int buf = 0;
#pragma unroll
    for (int s = 0; s < NSTEPS; ++s) {
        if (s + 1 < NSTEPS) {           // prefetch next tile into registers
            loadA((s + 1) * 32);
            loadB((s + 1) * 32);
        }
        // Operands: 32 contiguous bytes per lane -> 2x ds_load_b128 each.
        const v16bf av0 = *(const v16bf*)&Asw[buf][(wm * 2 + 0) * 2 + half][lr][0];
        const v16bf av1 = *(const v16bf*)&Asw[buf][(wm * 2 + 1) * 2 + half][lr][0];
        const v16bf bv0 = *(const v16bf*)&Bt[buf][wn * 32 + lr][half * 16];
        const v16bf bv1 = *(const v16bf*)&Bt[buf][wn * 32 + 16 + lr][half * 16];

        acc[0][0] = __builtin_amdgcn_wmma_f32_16x16x32_bf16(
            false, av0, false, bv0, (short)0, acc[0][0], false, false);
        acc[0][1] = __builtin_amdgcn_wmma_f32_16x16x32_bf16(
            false, av0, false, bv1, (short)0, acc[0][1], false, false);
        acc[1][0] = __builtin_amdgcn_wmma_f32_16x16x32_bf16(
            false, av1, false, bv0, (short)0, acc[1][0], false, false);
        acc[1][1] = __builtin_amdgcn_wmma_f32_16x16x32_bf16(
            false, av1, false, bv1, (short)0, acc[1][1], false, false);

        if (s + 1 < NSTEPS) {
            stage(buf ^ 1);             // write the *other* buffer
            __syncthreads();            // one barrier per K-step
        }
        buf ^= 1;
    }

    // ---- Epilogue ----------------------------------------------------------
#pragma unroll
    for (int ni = 0; ni < 2; ++ni) {
        const int col = n0 + wn * 32 + ni * 16 + lr;
        if (NBOUND && col >= N) continue;
        const float bval = bias ? bias[col] : 0.f;
#pragma unroll
        for (int mi = 0; mi < 2; ++mi) {
            const int rbase = m0 + wm * 32 + mi * 16 + half * 8;
#pragma unroll
            for (int r = 0; r < 8; ++r) {
                float v = acc[mi][ni][r] + bval;
                if (act == 1) v = softplus_f(v);
                C[(rbase + r) * ldc + col] = v;   // 32-bit offset
            }
        }
    }
}

// ---------------------------------------------------------------------------
// Causal (rev=0) / anti-causal (rev=1) depthwise conv (DC=4) + SiLU.
// xs is the low-512 slice of xr [ROWS_, 1024]; output dense [ROWS_, 512].
// ---------------------------------------------------------------------------
__global__ void conv_silu_kernel(const float* __restrict__ xr,
                                 const float* __restrict__ w,
                                 const float* __restrict__ bias,
                                 float* __restrict__ out, int rev) {
    const size_t idx = (size_t)blockIdx.x * blockDim.x + threadIdx.x;
    if (idx >= (size_t)ROWS_ * DI_) return;
    const int    d  = (int)(idx & (DI_ - 1));
    const size_t bl = idx >> 9;              // b*L + l
    const int    l  = (int)(bl & (L_ - 1));
    const size_t bb = bl >> 10;              // batch

    float acc = bias[d];
#pragma unroll
    for (int k = 0; k < 4; ++k) {
        const int off = 3 - k;               // w[3] hits the current sample
        const int ls  = rev ? (l + off) : (l - off);
        if (ls >= 0 && ls < L_)
            acc += w[d * 4 + k] * xr[((bb << 10) + ls) * (2 * DI_) + d];
    }
    out[idx] = silu_f(acc);
}

// ---------------------------------------------------------------------------
// Selective scan. One thread per (b, d) channel; h[16] lives in registers.
// B/C vectors (shared across d for a given (b,t)) broadcast through LDS.
// rev=1 runs time backwards (flip/unflip pair cancels).
// ---------------------------------------------------------------------------
__global__ void scan_kernel(const float* __restrict__ xdbl,   // [ROWS_, 48]
                            const float* __restrict__ delta,  // [ROWS_, 512]
                            const float* __restrict__ xsc,    // [ROWS_, 512]
                            const float* __restrict__ ALog,   // [512, 16]
                            const float* __restrict__ Dp,     // [512]
                            float* __restrict__ y, int rev) {
    const int b = blockIdx.x >> 1;
    const int d = ((blockIdx.x & 1) << 8) + threadIdx.x;   // 0..511

    __shared__ float BpS[DS_];
    __shared__ float CpS[DS_];

    float Aa[DS_];
#pragma unroll
    for (int n = 0; n < DS_; ++n) Aa[n] = -__expf(ALog[d * DS_ + n]);
    const float dpv = Dp[d];

    float h[DS_];
#pragma unroll
    for (int n = 0; n < DS_; ++n) h[n] = 0.f;

    for (int t = 0; t < L_; ++t) {
        const int    tt  = rev ? (L_ - 1 - t) : t;
        const size_t row = (size_t)b * L_ + tt;

        if (threadIdx.x < 32) {
            const int n = threadIdx.x & 15;
            const float v = xdbl[row * XDBL_ + DTR_ + ((threadIdx.x & 16) ? DS_ : 0) + n];
            if (threadIdx.x < 16) BpS[n] = v; else CpS[n] = v;
        }
        __syncthreads();

        const float dl = delta[row * DI_ + d];
        const float xv = xsc[row * DI_ + d];
        const float dx = dl * xv;
        float acc = 0.f;
#pragma unroll
        for (int n = 0; n < DS_; ++n) {
            const float dA = __expf(dl * Aa[n]);
            h[n] = dA * h[n] + dx * BpS[n];
            acc += h[n] * CpS[n];
        }
        y[row * DI_ + d] = acc + dpv * xv;
        __syncthreads();
    }
}

// ---------------------------------------------------------------------------
// y *= silu(res), res = high-512 slice of xr.
// ---------------------------------------------------------------------------
__global__ void silu_mul_kernel(float* __restrict__ y,
                                const float* __restrict__ xr) {
    const size_t idx = (size_t)blockIdx.x * blockDim.x + threadIdx.x;
    if (idx >= (size_t)ROWS_ * DI_) return;
    const int    d  = (int)(idx & (DI_ - 1));
    const size_t bl = idx >> 9;
    const float  r  = xr[bl * (2 * DI_) + DI_ + d];
    y[idx] *= silu_f(r);
}

// ---------------------------------------------------------------------------
// out = x + fused
// ---------------------------------------------------------------------------
__global__ void residual_add_kernel(const float* __restrict__ x,
                                    const float* __restrict__ fused,
                                    float* __restrict__ out) {
    const size_t idx = (size_t)blockIdx.x * blockDim.x + threadIdx.x;
    if (idx >= (size_t)ROWS_ * DM_) return;
    out[idx] = x[idx] + fused[idx];
}

// ---------------------------------------------------------------------------
extern "C" void kernel_launch(void* const* d_in, const int* in_sizes, int n_in,
                              void* d_out, int out_size, void* d_ws, size_t ws_size,
                              hipStream_t stream) {
    (void)in_sizes; (void)n_in; (void)out_size; (void)ws_size;

    const float* x      = (const float*)d_in[0];
    const float* norm_g = (const float*)d_in[1];
    const float* norm_b = (const float*)d_in[2];
    // per-direction weights: f at base 3, b at base 12
    // [inW, convW, convB, xW, dtW, dtB, ALog, Dp, outW]
    const float* fusW = (const float*)d_in[21];
    const float* fusB = (const float*)d_in[22];
    float* out = (float*)d_out;

    // Workspace carve-up (fp32 elements), 256B aligned.
    char* ws = (char*)d_ws;
    size_t cur = 0;
    auto carve = [&](size_t elems) {
        float* p = (float*)(ws + cur);
        cur += ((elems * sizeof(float)) + 255) & ~(size_t)255;
        return p;
    };
    float* xn    = carve((size_t)ROWS_ * DM_);        // 8 MB
    float* xr    = carve((size_t)ROWS_ * 2 * DI_);    // 32 MB
    float* xsc   = carve((size_t)ROWS_ * DI_);        // 16 MB
    float* xdbl  = carve((size_t)ROWS_ * XDBL_);      // 1.5 MB
    float* delta = carve((size_t)ROWS_ * DI_);        // 16 MB
    float* ybuf  = carve((size_t)ROWS_ * DI_);        // 16 MB
    float* ycat  = carve((size_t)ROWS_ * 2 * DM_);    // 16 MB
    float* fused = carve((size_t)ROWS_ * DM_);        // 8 MB

    const dim3 blk(256);

    // 1) LayerNorm
    ln_kernel<<<dim3(ROWS_ / 8), blk, 0, stream>>>(x, norm_g, norm_b, xn);

    // 2) Both Mamba directions
    for (int dir = 0; dir < 2; ++dir) {
        const int base = 3 + dir * 9;
        const float* inW   = (const float*)d_in[base + 0];  // [256,1024]
        const float* convW = (const float*)d_in[base + 1];  // [512,4]
        const float* convB = (const float*)d_in[base + 2];  // [512]
        const float* xW    = (const float*)d_in[base + 3];  // [512,48]
        const float* dtW   = (const float*)d_in[base + 4];  // [16,512]
        const float* dtB   = (const float*)d_in[base + 5];  // [512]
        const float* ALog  = (const float*)d_in[base + 6];  // [512,16]
        const float* Dp    = (const float*)d_in[base + 7];  // [512]
        const float* outW  = (const float*)d_in[base + 8];  // [512,256]

        // xr = xn @ inW                  [8192,256] x [256,1024]
        gemm_wmma_bf16<DM_, false><<<dim3(ROWS_ / 128, (2 * DI_) / 64), blk, 0, stream>>>(
            xn, DM_, inW, 2 * DI_, xr, 2 * DI_, ROWS_, 2 * DI_, nullptr, 0);

        // xsc = silu(dwconv(xs))         (xs = low half of xr)
        conv_silu_kernel<<<dim3((ROWS_ * DI_) / 256), blk, 0, stream>>>(
            xr, convW, convB, xsc, dir);

        // xdbl = xsc @ xW                [8192,512] x [512,48]  (ragged N)
        gemm_wmma_bf16<DI_, true><<<dim3(ROWS_ / 128, 1), blk, 0, stream>>>(
            xsc, DI_, xW, XDBL_, xdbl, XDBL_, ROWS_, XDBL_, nullptr, 0);

        // delta = softplus(dlt @ dtW + dtB)   K=16 zero-padded to 32
        gemm_wmma_bf16<DTR_, false><<<dim3(ROWS_ / 128, DI_ / 64), blk, 0, stream>>>(
            xdbl, XDBL_, dtW, DI_, delta, DI_, ROWS_, DI_, dtB, 1);

        // selective scan -> ybuf
        scan_kernel<<<dim3(B_ * 2), blk, 0, stream>>>(
            xdbl, delta, xsc, ALog, Dp, ybuf, dir);

        // ybuf *= silu(res)              (res = high half of xr)
        silu_mul_kernel<<<dim3((ROWS_ * DI_) / 256), blk, 0, stream>>>(ybuf, xr);

        // ycat[:, dir*256 : dir*256+256] = ybuf @ outW   [8192,512]x[512,256]
        gemm_wmma_bf16<DI_, false><<<dim3(ROWS_ / 128, DM_ / 64), blk, 0, stream>>>(
            ybuf, DI_, outW, DM_, ycat + dir * DM_, 2 * DM_, ROWS_, DM_, nullptr, 0);
    }

    // 3) fused = ycat @ fusW + fusB     [8192,512] x [512,256]
    gemm_wmma_bf16<DI_, false><<<dim3(ROWS_ / 128, DM_ / 64), blk, 0, stream>>>(
        ycat, 2 * DM_, fusW, DM_, fused, DM_, ROWS_, DM_, fusB, 0);

    // 4) out = x + fused
    residual_add_kernel<<<dim3((ROWS_ * DM_) / 256), blk, 0, stream>>>(
        x, fused, out);
}
